// BERelativeSelfMultiheadAttn_4544075399696
// MI455X (gfx1250) — matrix operand order repaired and checked by reading, not compiled, not run on hardware
//
#include <hip/hip_runtime.h>

// BERelativeSelfMultiheadAttn for MI455X (gfx1250, wave32, WMMA).
// Fixed problem size from the reference: T=1024, B=8, H=1024, heads=16, hd=64, Lr=1024.
// Pipeline: bf16 convert+scale -> WMMA GEMM (QKV) -> WMMA GEMM (pos) ->
//           fused flash attention with rel_shift folded in -> WMMA GEMM (output).
// CDNA5 paths: v_wmma_f32_16x16x32_bf16, global_load_async_to_lds_b128 (+ASYNCcnt),
//              ds_bpermute rel-shift select, global_prefetch_b8.

#define DEVI __device__ __forceinline__

typedef __attribute__((ext_vector_type(16))) __bf16 v16bf;
typedef __attribute__((ext_vector_type(8)))  float  v8f;

union Frag { uint4 u[2]; v16bf v; };

DEVI v8f wmma_bf16(v16bf a, v16bf b, v8f c) {
  // D = A(16x32 bf16) * B(32x16 bf16) + C(16x16 f32)
  return __builtin_amdgcn_wmma_f32_16x16x32_bf16(false, a, false, b, (short)0, c,
                                                 false, false);
}

// Async global->LDS 16-byte copy (CDNA5 GLOBAL_LOAD_ASYNC_TO_LDS_B128, ASYNCcnt).
// LDS operand is the 32-bit LDS byte address (low 32 bits of the flat address).
DEVI void async_ld_b128(void* lds, const void* g) {
  unsigned l = (unsigned)(size_t)lds;
  unsigned long long a = (unsigned long long)(size_t)g;
  asm volatile("global_load_async_to_lds_b128 %0, %1, off"
               :: "v"(l), "v"(a) : "memory");
}
DEVI void wait_async0() { asm volatile("s_wait_asynccnt 0x0" ::: "memory"); }

// A-fragment (16x32 bf16), row-major source with leading dim `ld`.
// Lane L: m=L&15, Ks = 16*(L>>4) + 0..15 (contiguous).
DEVI v16bf a_frag(const __bf16* p0, int ld, int m0, int k0) {
  int lane = threadIdx.x & 31;
  const __bf16* p = p0 + (size_t)(m0 + (lane & 15)) * ld + k0 + ((lane >> 4) << 4);
  Frag f;
  f.u[0] = *(const uint4*)p;
  f.u[1] = *(const uint4*)(p + 8);
  return f.v;
}

// B-fragment (32x16 bf16) from an N-major ("NT") source: row n holds the K-dim.
// Lane L holds column n = L&15, Ks = 16*(L>>4) + 0..15 (contiguous).
DEVI v16bf b_frag(const __bf16* p0, int ld, int n0, int k0) {
  int lane = threadIdx.x & 31;
  const __bf16* p = p0 + (size_t)(n0 + (lane & 15)) * ld + k0 + ((lane >> 4) << 4);
  Frag f;
  f.u[0] = *(const uint4*)p;
  f.u[1] = *(const uint4*)(p + 8);
  return f.v;
}

// B-fragment from Rk with row range guard (rows outside [0,1024) read as zero,
// which also implements the k>q zero region of rel_shift).
DEVI v16bf rk_frag(const __bf16* Rkn, int rb, int j) {
  int lane = threadIdx.x & 31;
  int r = rb + (lane & 15);
  Frag f;
  if ((unsigned)r < 1024u) {
    const __bf16* p = Rkn + (size_t)r * 64 + (j << 5) + ((lane >> 4) << 4);
    f.u[0] = *(const uint4*)p;
    f.u[1] = *(const uint4*)(p + 8);
  } else {
    f.u[0] = uint4{0u, 0u, 0u, 0u};
    f.u[1] = uint4{0u, 0u, 0u, 0u};
  }
  return f.v;
}

// ---------------- elementwise prep ----------------

__global__ __launch_bounds__(256) void prep_xr(const float* __restrict__ x,
                                               const float* __restrict__ r,
                                               __bf16* __restrict__ o) {
  int i = blockIdx.x * 256 + threadIdx.x;          // (t, b, h), T*B*H elems
  int h = i & 1023, b = (i >> 10) & 7;
  o[i] = (__bf16)(x[i] * r[(b << 10) + h]);
}

__global__ __launch_bounds__(256) void prep_pr(const float* __restrict__ pos,
                                               const float* __restrict__ r,
                                               __bf16* __restrict__ o) {
  int i = blockIdx.x * 256 + threadIdx.x;          // layout (b, l, h)
  int h = i & 1023, l = (i >> 10) & 1023, b = i >> 20;
  o[i] = (__bf16)(pos[(l << 10) + h] * r[(b << 10) + h]);
}

__global__ __launch_bounds__(256) void f32_to_bf16(const float* __restrict__ in,
                                                   __bf16* __restrict__ o, int n) {
  int i = blockIdx.x * 256 + threadIdx.x;
  if (i < n) o[i] = (__bf16)in[i];
}

// ---------------- GEMM epilogues ----------------

struct EpiQKV {
  const float *s_i, *bi, *rwb, *rrb;
  __bf16 *Qw, *Qr, *Kb, *Vb;
  float scale;
  __device__ void operator()(int m, int o, float v) const {
    int b = m & 7, t = m >> 3;                       // m = t*B + b
    float val = v * s_i[b * 3072 + o] + bi[o];
    int hh = o / 192, rem = o - hh * 192;            // 3H -> (heads, 3, hd)
    int c = rem >> 6, d = rem & 63;
    size_t idx = (((size_t)((b << 4) | hh)) * 1024 + t) * 64 + d;
    if (c == 0) {
      int bo = (hh << 6) + d;
      Qw[idx] = (__bf16)((val + rwb[bo]) * scale);   // fold 1/sqrt(hd)
      Qr[idx] = (__bf16)((val + rrb[bo]) * scale);
    } else if (c == 1) {
      Kb[idx] = (__bf16)val;
    } else {
      Vb[idx] = (__bf16)val;
    }
  }
};

struct EpiRK {
  const float *s_p, *pb;
  __bf16* Rk;
  __device__ void operator()(int m, int o, float v) const {
    int b = m >> 10, l = m & 1023;                   // m = b*Lr + l
    float val = v * s_p[b * 1024 + o] + pb[o];
    int hh = o >> 6, d = o & 63;
    Rk[(((size_t)((b << 4) | hh)) * 1024 + l) * 64 + d] = (__bf16)val;
  }
};

struct EpiOut {
  const float* ob;
  float* out;
  __device__ void operator()(int m, int o, float v) const {
    out[(size_t)m * 1024 + o] = v + ob[o];
  }
};

// ---------------- tiled NT GEMM: C[M,N] = A[M,K] * Bw[N,K]^T ----------------
// Block tile 128x128, 8 waves (2 M x 4 N), wave tile 64x32 = 4x2 WMMA tiles.
// Double-buffered LDS with async global->LDS staging; next+1 tile prefetched.

template <class Epi>
__global__ __launch_bounds__(256) void gemm_nt(const __bf16* __restrict__ A,
                                               const __bf16* __restrict__ Bw,
                                               int K, Epi epi) {
  __shared__ __align__(16) __bf16 sA[2][128][32];
  __shared__ __align__(16) __bf16 sB[2][128][32];
  const int tid = threadIdx.x, lane = tid & 31, wave = tid >> 5;
  const int m_blk = blockIdx.y << 7, n_blk = blockIdx.x << 7;
  const int wm = (wave & 1) << 6, wn = (wave >> 1) << 5;
  const v8f vz = {0.f, 0.f, 0.f, 0.f, 0.f, 0.f, 0.f, 0.f};
  v8f acc[4][2];
#pragma unroll
  for (int i = 0; i < 4; ++i)
#pragma unroll
    for (int j = 0; j < 2; ++j) acc[i][j] = vz;

  const int r = tid >> 1, c = (tid & 1) << 4;
  const __bf16* gA = A + (size_t)(m_blk + r) * K + c;
  const __bf16* gB = Bw + (size_t)(n_blk + r) * K + c;

  // stage one 128x32 panel pair into buffer `buf` via async loads
  auto stage = [&](int buf, int k0) {
    async_ld_b128(&sA[buf][r][c],     gA + k0);
    async_ld_b128(&sA[buf][r][c + 8], gA + k0 + 8);
    async_ld_b128(&sB[buf][r][c],     gB + k0);
    async_ld_b128(&sB[buf][r][c + 8], gB + k0 + 8);
  };

  stage(0, 0);
  wait_async0();
  __syncthreads();

  const int nk = K >> 5;
  for (int kt = 0; kt < nk; ++kt) {
    const int cur = kt & 1;
    if (kt + 1 < nk) stage(cur ^ 1, (kt + 1) << 5);
    if (kt + 2 < nk) {
      __builtin_prefetch(gA + ((kt + 2) << 5), 0, 0);
      __builtin_prefetch(gB + ((kt + 2) << 5), 0, 0);
    }
    v16bf af[4], bfv[2];
#pragma unroll
    for (int i = 0; i < 4; ++i) af[i] = a_frag(&sA[cur][0][0], 32, wm + (i << 4), 0);
#pragma unroll
    for (int j = 0; j < 2; ++j) bfv[j] = b_frag(&sB[cur][0][0], 32, wn + (j << 4), 0);
#pragma unroll
    for (int i = 0; i < 4; ++i)
#pragma unroll
      for (int j = 0; j < 2; ++j) acc[i][j] = wmma_bf16(af[i], bfv[j], acc[i][j]);
    wait_async0();        // this wave's stage of the next buffer is done
    __syncthreads();      // everyone done reading cur & staging nxt
  }
  const int half = lane >> 4, n16 = lane & 15;
#pragma unroll
  for (int i = 0; i < 4; ++i)
#pragma unroll
    for (int j = 0; j < 2; ++j)
#pragma unroll
      for (int v = 0; v < 8; ++v)
        epi(m_blk + wm + (i << 4) + (half << 3) + v,
            n_blk + wn + (j << 4) + n16, acc[i][j][v]);
}

// ---------------- fused attention ----------------
// Block = (head-batch n, 128 q rows); 8 waves x 16 q rows each.
// Per 32-wide k step: S = Qw*K^T + rel_shift(Qr*Rk^T); online softmax; O += P*V.
// rel_shift fused: BD_shift[q,k] = BD[q, k-q+T-1]; per 16x16 tile this is a
// diagonal band spanning two adjacent BD column tiles, selected per row via
// ds_bpermute (__shfl width 16) on the WMMA accumulators.

__global__ __launch_bounds__(256) void attn_kernel(const __bf16* __restrict__ Qw,
                                                   const __bf16* __restrict__ Qr,
                                                   const __bf16* __restrict__ Kb,
                                                   const __bf16* __restrict__ Vb,
                                                   const __bf16* __restrict__ Rk,
                                                   __bf16* __restrict__ Ctx) {
  const int T = 1024;
  __shared__ __align__(16) __bf16 sK[32][64];
  __shared__ __align__(16) __bf16 sVt[64][32];     // V transposed: [d][k]
  __shared__ __align__(16) __bf16 sP[8][16][32];   // per-wave P tile
  const int tid = threadIdx.x, lane = tid & 31, wave = tid >> 5;
  const int half = lane >> 4, n16 = lane & 15;
  const int n = blockIdx.x;                        // 0..127 (b*16 + head)
  const int qt = (blockIdx.y << 7) + (wave << 4);
  const int b = n >> 4, hh = n & 15;
  const __bf16* Qwn = Qw + (size_t)n * T * 64;
  const __bf16* Qrn = Qr + (size_t)n * T * 64;
  const __bf16* Kn  = Kb + (size_t)n * T * 64;
  const __bf16* Vn  = Vb + (size_t)n * T * 64;
  const __bf16* Rkn = Rk + (size_t)n * T * 64;

  const v16bf qw0 = a_frag(Qwn, 64, qt, 0), qw1 = a_frag(Qwn, 64, qt, 32);
  const v16bf qr0 = a_frag(Qrn, 64, qt, 0), qr1 = a_frag(Qrn, 64, qt, 32);

  const v8f vz = {0.f, 0.f, 0.f, 0.f, 0.f, 0.f, 0.f, 0.f};
  v8f o_acc[4];
  float m_run[8], l_run[8];
#pragma unroll
  for (int t = 0; t < 4; ++t) o_acc[t] = vz;
#pragma unroll
  for (int v = 0; v < 8; ++v) { m_run[v] = -1e30f; l_run[v] = 0.f; }

  const int sr = tid >> 3;            // 0..31 : k row within tile
  const int sc = (tid & 7) << 3;      // 0..56 : d column base

  for (int k0 = 0; k0 < T; k0 += 32) {
    __syncthreads();
    // K tile: async global->LDS (ASYNCcnt); V tile: VGPR transpose into sVt.
    async_ld_b128(&sK[sr][sc], Kn + (size_t)(k0 + sr) * 64 + sc);
    {
      uint4 vv = *(const uint4*)(Vn + (size_t)(k0 + sr) * 64 + sc);
      const __bf16* ve = (const __bf16*)&vv;
#pragma unroll
      for (int e = 0; e < 8; ++e) sVt[sc + e][sr] = ve[e];
    }
    wait_async0();
    __syncthreads();

    // AC tiles (k columns k0..k0+15, k0+16..k0+31)
    v8f ac[2];
#pragma unroll
    for (int s = 0; s < 2; ++s) {
      v8f a = vz;
      a = wmma_bf16(qw0, b_frag(&sK[0][0], 64, s << 4, 0), a);
      a = wmma_bf16(qw1, b_frag(&sK[0][0], 64, s << 4, 32), a);
      ac[s] = a;
    }

    // BD tiles at Rk row bases C0-16, C0, C0+16 (C0 = k0 - qt + T-1)
    const int C0 = k0 - qt + (T - 1);
    v8f bd[3];
#pragma unroll
    for (int s = 0; s < 3; ++s) {
      int base = C0 + ((s - 1) << 4);
      v8f a = vz;
      a = wmma_bf16(qr0, rk_frag(Rkn, base, 0), a);
      a = wmma_bf16(qr1, rk_frag(Rkn, base, 1), a);
      bd[s] = a;
    }

    // rel-shift diagonal select + scores
    float S0[8], S1[8];
#pragma unroll
    for (int v = 0; v < 8; ++v) {
      int m = (half << 3) + v;
      int diff = n16 - m;
      int src = diff & 15;
      float r1 = __shfl(bd[1][v], src, 16);   // base C0
      float l1 = __shfl(bd[0][v], src, 16);   // base C0-16
      float r2 = __shfl(bd[2][v], src, 16);   // base C0+16
      S0[v] = ac[0][v] + (diff >= 0 ? r1 : l1);
      S1[v] = ac[1][v] + (diff >= 0 ? r2 : r1);  // L of tile1 == R of tile0
    }

    // online softmax (row stats live in 16-lane halves by C-layout)
#pragma unroll
    for (int v = 0; v < 8; ++v) {
      float mx = fmaxf(S0[v], S1[v]);
#pragma unroll
      for (int off = 1; off < 16; off <<= 1) mx = fmaxf(mx, __shfl_xor(mx, off, 16));
      float mnew = fmaxf(m_run[v], mx);
      float alpha = __expf(m_run[v] - mnew);
      float p0 = __expf(S0[v] - mnew);
      float p1 = __expf(S1[v] - mnew);
      float sum = p0 + p1;
#pragma unroll
      for (int off = 1; off < 16; off <<= 1) sum += __shfl_xor(sum, off, 16);
      l_run[v] = l_run[v] * alpha + sum;
      m_run[v] = mnew;
#pragma unroll
      for (int t = 0; t < 4; ++t) o_acc[t][v] *= alpha;
      int m = (half << 3) + v;
      sP[wave][m][n16]      = (__bf16)p0;
      sP[wave][m][16 + n16] = (__bf16)p1;
    }

    // O += P @ V  (P reshaped C-layout -> A-fragment via LDS, V from sVt)
    v16bf pf = a_frag(&sP[wave][0][0], 32, 0, 0);
#pragma unroll
    for (int t = 0; t < 4; ++t)
      o_acc[t] = wmma_bf16(pf, b_frag(&sVt[0][0], 32, t << 4, 0), o_acc[t]);
  }

  // Ctx[(q*B + b)][hh*64 + d] = O / l
#pragma unroll
  for (int t = 0; t < 4; ++t)
#pragma unroll
    for (int v = 0; v < 8; ++v) {
      int q = qt + (half << 3) + v;
      int d = (t << 4) + n16;
      float val = o_acc[t][v] / l_run[v];
      Ctx[((size_t)q * 8 + b) * 1024 + (hh << 6) + d] = (__bf16)val;
    }
}

// ---------------- host launcher ----------------

extern "C" void kernel_launch(void* const* d_in, const int* in_sizes, int n_in,
                              void* d_out, int out_size, void* d_ws, size_t ws_size,
                              hipStream_t stream) {
  const float* inputs = (const float*)d_in[0];
  const float* pos    = (const float*)d_in[1];
  const float* Wi     = (const float*)d_in[2];
  const float* bi     = (const float*)d_in[3];
  const float* Wp     = (const float*)d_in[4];
  const float* pb     = (const float*)d_in[5];
  const float* Wo     = (const float*)d_in[6];
  const float* ob     = (const float*)d_in[7];
  const float* r_i    = (const float*)d_in[8];
  const float* s_i    = (const float*)d_in[9];
  const float* r_p    = (const float*)d_in[10];
  const float* s_p    = (const float*)d_in[11];
  const float* rwb    = (const float*)d_in[12];
  const float* rrb    = (const float*)d_in[13];
  float* out = (float*)d_out;

  char* ws = (char*)d_ws;
  const size_t MB = 1ull << 20;
  __bf16* Xr  = (__bf16*)(ws + 0 * MB);     // (T*B, H)       16MB
  __bf16* Pr  = (__bf16*)(ws + 16 * MB);    // (B, Lr, H)     16MB
  __bf16* Wib = (__bf16*)(ws + 32 * MB);    // (3H, H)         6MB
  __bf16* Wpb = (__bf16*)(ws + 38 * MB);    // (H, H)          2MB
  __bf16* Wob = (__bf16*)(ws + 40 * MB);    // (H, H)          2MB
  __bf16* Qw  = (__bf16*)(ws + 42 * MB);    // (128, T, 64)   16MB
  __bf16* Qr  = (__bf16*)(ws + 58 * MB);
  __bf16* Kb  = (__bf16*)(ws + 74 * MB);
  __bf16* Vb  = (__bf16*)(ws + 90 * MB);
  __bf16* Rk  = (__bf16*)(ws + 106 * MB);
  __bf16* Ctx = (__bf16*)(ws + 122 * MB);   // (T*B, H)       16MB -> 138MB total

  prep_xr<<<8388608 / 256, 256, 0, stream>>>(inputs, r_i, Xr);
  prep_pr<<<8388608 / 256, 256, 0, stream>>>(pos, r_p, Pr);
  f32_to_bf16<<<3145728 / 256, 256, 0, stream>>>(Wi, Wib, 3145728);
  f32_to_bf16<<<1048576 / 256, 256, 0, stream>>>(Wp, Wpb, 1048576);
  f32_to_bf16<<<1048576 / 256, 256, 0, stream>>>(Wo, Wob, 1048576);

  EpiQKV e1{s_i, bi, rwb, rrb, Qw, Qr, Kb, Vb, 0.125f};
  gemm_nt<EpiQKV><<<dim3(3072 / 128, 8192 / 128), 256, 0, stream>>>(Xr, Wib, 1024, e1);

  EpiRK e2{s_p, pb, Rk};
  gemm_nt<EpiRK><<<dim3(1024 / 128, 8192 / 128), 256, 0, stream>>>(Pr, Wpb, 1024, e2);

  attn_kernel<<<dim3(128, 8), 256, 0, stream>>>(Qw, Qr, Kb, Vb, Rk, Ctx);

  EpiOut e3{ob, out};
  gemm_nt<EpiOut><<<dim3(1024 / 128, 8192 / 128), 256, 0, stream>>>(Ctx, Wob, 1024, e3);

  (void)in_sizes; (void)n_in; (void)out_size; (void)ws_size;
}
